// RankAugmentedLinearAttention_541165879756
// MI455X (gfx1250) — matrix operand order repaired and checked by reading, not compiled, not run on hardware
//
#include <hip/hip_runtime.h>
#include <hip/hip_bf16.h>

// ---------------------------------------------------------------------------
// Rank-augmented linear attention for MI455X (gfx1250, wave32, WMMA).
// All heavy math on v_wmma_f32_16x16x32_bf16; LoRA folded into weights;
// B matrices pre-transposed so LDS staging is pure b128 traffic.
// ---------------------------------------------------------------------------

typedef __attribute__((ext_vector_type(16))) __bf16 v16bf;
typedef __attribute__((ext_vector_type(8)))  __bf16 bf16x8;
typedef __attribute__((ext_vector_type(4)))  __bf16 bf16x4;
typedef __attribute__((ext_vector_type(8)))  float  v8f;
typedef __attribute__((ext_vector_type(4)))  float  f32x4;

#define E_DIM  1024
#define H_NUM  16
#define D_DIM  64
#define B_NUM  4
#define S_LEN  4096
#define R_RANK 8
#define BKPAD  40   // 32 K-elements + 8 pad (bank-conflict mitigation)

// ---------------------------------------------------------------------------
// WMMA fragment helpers (layouts per CDNA5 ISA 7.12.2, wave32)
// ---------------------------------------------------------------------------
static __device__ __forceinline__ v16bf make_frag(bf16x8 lo, bf16x8 hi) {
  return __builtin_shufflevector(lo, hi, 0,1,2,3,4,5,6,7,8,9,10,11,12,13,14,15);
}

// A fragment (16x32 bf16). LDS layout: [row][k], row stride BKPAD.
// lanes 0-15: row M=lane, K = 0..7 then 16..23 ; lanes 16-31: K = 8..15 then 24..31
static __device__ __forceinline__ v16bf load_a_frag(const __bf16* lds, int rowBase, int lane) {
  const int row = rowBase + (lane & 15);
  const int kb  = (lane >> 4) << 3;               // 0 or 8
  const __bf16* p = lds + row * BKPAD;
  bf16x8 lo = *(const bf16x8*)(p + kb);
  bf16x8 hi = *(const bf16x8*)(p + kb + 16);
  return make_frag(lo, hi);
}

// B fragment (32x16 bf16) from K-transposed LDS: [col][k], stride BKPAD.
// lanes 0-15: col N=lane, K = 0..7 then 16..23 ; lanes 16-31: K = 8..15 then 24..31
static __device__ __forceinline__ v16bf load_b_frag(const __bf16* lds, int colBase, int lane) {
  const int col = colBase + (lane & 15);
  const int kb  = (lane >> 4) << 3;               // 0 or 8
  const __bf16* p = lds + col * BKPAD;
  bf16x8 lo = *(const bf16x8*)(p + kb);
  bf16x8 hi = *(const bf16x8*)(p + kb + 16);
  return make_frag(lo, hi);
}

// ---------------------------------------------------------------------------
// Generic bf16 GEMM:  C[z] = A[z] (MxK) @ B[z] (KxN) + bias, B given TRANSPOSED
// as Bt (NxK, row stride ldbt). Macro tile 128xBN, BK=32, 256 threads = 8 waves;
// each wave owns 32 rows x BN/2 cols (JT = BN/32 WMMA col-tiles).
// z (blockIdx.z) decodes to (batch, head) offsets for per-head GEMMs.
// ---------------------------------------------------------------------------
template <int BN, typename OutT>
__global__ __launch_bounds__(256)
void gemm_bf16_wmma(const __bf16* __restrict__ A, int lda,
                    const __bf16* __restrict__ Bt, int ldbt,
                    OutT* __restrict__ C, int ldc,
                    const float* __restrict__ bias,
                    int K,
                    long sAb, long sAh, long sBz, long sCb, long sCh)
{
  constexpr int JT = BN / 32;               // 16-col tiles per wave
  __shared__ __bf16 Alds[128 * BKPAD];
  __shared__ __bf16 Btlds[BN * BKPAD];

  const int tid  = threadIdx.x;
  const int lane = tid & 31;
  const int wave = tid >> 5;
  const int n0   = blockIdx.x * BN;
  const int m0   = blockIdx.y * 128;
  const int z    = blockIdx.z;

  A  += (long)(z >> 4) * sAb + (long)(z & 15) * sAh;
  Bt += (long)z * sBz;
  C  += (long)(z >> 4) * sCb + (long)(z & 15) * sCh;

  const int wr = wave >> 1;   // 0..3 : 32-row group
  const int wc = wave & 1;    // 0..1 : BN/2-col group

  v8f acc[2][JT] = {};

  // cooperative load mapping
  const int aRow = tid >> 2;           // 0..63 (+64 for second half)
  const int aCol = (tid & 3) << 3;     // 0,8,16,24
  const int bRow = tid >> 2;           // 0..63 (Bt rows = N)
  const int bCol = (tid & 3) << 3;     // 0,8,16,24 (K direction)

  for (int k0 = 0; k0 < K; k0 += 32) {
    // stage A (128x32) : b128 in, b128 out
    {
      bf16x8 g0 = *(const bf16x8*)(A + (size_t)(m0 + aRow) * lda + k0 + aCol);
      bf16x8 g1 = *(const bf16x8*)(A + (size_t)(m0 + aRow + 64) * lda + k0 + aCol);
      *(bf16x8*)(Alds + aRow * BKPAD + aCol)        = g0;
      *(bf16x8*)(Alds + (aRow + 64) * BKPAD + aCol) = g1;
    }
    // stage Bt (BNx32) : already K-major, pure b128 traffic
    #pragma unroll
    for (int nb = 0; nb < BN; nb += 64) {
      bf16x8 g = *(const bf16x8*)(Bt + (size_t)(n0 + nb + bRow) * ldbt + k0 + bCol);
      *(bf16x8*)(Btlds + (nb + bRow) * BKPAD + bCol) = g;
    }
    if (k0 + 32 < K) {  // hint next K tile -> global_prefetch_b8
      __builtin_prefetch(A + (size_t)(m0 + aRow) * lda + (k0 + 32) + aCol, 0, 0);
      __builtin_prefetch(Bt + (size_t)(n0 + bRow) * ldbt + (k0 + 32) + bCol, 0, 0);
    }
    __syncthreads();

    v16bf a0 = load_a_frag(Alds, wr * 32,      lane);
    v16bf a1 = load_a_frag(Alds, wr * 32 + 16, lane);
    #pragma unroll
    for (int j = 0; j < JT; ++j) {
      v16bf b = load_b_frag(Btlds, wc * (BN / 2) + j * 16, lane);
      acc[0][j] = __builtin_amdgcn_wmma_f32_16x16x32_bf16(false, a0, false, b, (short)0, acc[0][j], false, false);
      acc[1][j] = __builtin_amdgcn_wmma_f32_16x16x32_bf16(false, a1, false, b, (short)0, acc[1][j], false, false);
    }
    __syncthreads();
  }

  // epilogue: f32 16x16 C layout -> lane = N, VGPR r = M (M+8 for upper lanes)
  const int col   = lane & 15;
  const int rbase = (lane >> 4) << 3;
  #pragma unroll
  for (int i = 0; i < 2; ++i) {
    #pragma unroll
    for (int j = 0; j < JT; ++j) {
      const int gm0 = m0 + wr * 32 + i * 16 + rbase;
      const int gn  = n0 + wc * (BN / 2) + j * 16 + col;
      const float bv = bias ? bias[gn] : 0.0f;
      #pragma unroll
      for (int r = 0; r < 8; ++r) {
        float v = acc[i][j][r] + bv;
        C[(size_t)(gm0 + r) * ldc + gn] = (OutT)v;
      }
    }
  }
}

// ---------------------------------------------------------------------------
// ctx[b,h] = k_norm[b,h]^T (DxS) @ v[b,h] (SxD) : 64x64, K=4096, 8-way K-split
// with f32 global atomicAdd reduction. 8 waves, 2 WMMA tiles per wave.
// (Inputs are row-major [s][d]; transpose-staged through LDS.)
// ---------------------------------------------------------------------------
__global__ __launch_bounds__(256)
void ctx_gemm_wmma(const __bf16* __restrict__ Kn, const __bf16* __restrict__ V,
                   float* __restrict__ Ctx)
{
  __shared__ __bf16 Alds[64 * BKPAD];
  __shared__ __bf16 Btlds[64 * BKPAD];

  const int tid  = threadIdx.x;
  const int lane = tid & 31;
  const int wave = tid >> 5;
  const int ks   = blockIdx.x;          // 0..7 K-split
  const int bh   = blockIdx.y;          // 0..63
  const int b    = bh >> 4, h = bh & 15;

  const size_t base = (size_t)b * S_LEN * E_DIM + (size_t)h * D_DIM;
  const int sBeg = ks * (S_LEN / 8);

  const int mt  = wave >> 1;            // 0..3
  const int nt0 = (wave & 1) * 2;       // 0 or 2

  v8f acc0 = {}, acc1 = {};

  const int sl = tid >> 3;              // 0..31 local s
  const int d8 = (tid & 7) << 3;        // 0..56

  for (int s0 = sBeg; s0 < sBeg + (S_LEN / 8); s0 += 32) {
    bf16x8 ga = *(const bf16x8*)(Kn + base + (size_t)(s0 + sl) * E_DIM + d8);
    bf16x8 gb = *(const bf16x8*)(V  + base + (size_t)(s0 + sl) * E_DIM + d8);
    #pragma unroll
    for (int i = 0; i < 8; ++i) {       // transpose into [d][s] / [e][s]
      Alds[(d8 + i) * BKPAD + sl]  = ga[i];
      Btlds[(d8 + i) * BKPAD + sl] = gb[i];
    }
    __syncthreads();

    v16bf a  = load_a_frag(Alds,  mt  * 16, lane);
    v16bf b0 = load_b_frag(Btlds, nt0 * 16, lane);
    v16bf b1 = load_b_frag(Btlds, nt0 * 16 + 16, lane);
    acc0 = __builtin_amdgcn_wmma_f32_16x16x32_bf16(false, a, false, b0, (short)0, acc0, false, false);
    acc1 = __builtin_amdgcn_wmma_f32_16x16x32_bf16(false, a, false, b1, (short)0, acc1, false, false);
    __syncthreads();
  }

  const int col   = lane & 15;
  const int rbase = (lane >> 4) << 3;
  float* dst = Ctx + (size_t)bh * D_DIM * D_DIM;
  #pragma unroll
  for (int r = 0; r < 8; ++r) {
    const int gm = mt * 16 + rbase + r;
    atomicAdd(dst + (size_t)gm * D_DIM + (nt0 * 16 + col),       acc0[r]);
    atomicAdd(dst + (size_t)gm * D_DIM + ((nt0 + 1) * 16 + col), acc1[r]);
  }
}

// ---------------------------------------------------------------------------
// small helper kernels
// ---------------------------------------------------------------------------

// Fold rank-8 LoRA into the dense weight and store TRANSPOSED bf16: Wt[n][k].
// (Strided f32 reads are L2-resident: the whole matrix is 4MB.)
__global__ void fold_lora_t_k(const float* __restrict__ W,
                              const float* __restrict__ la,
                              const float* __restrict__ lb,
                              __bf16* __restrict__ outT)
{
  const int idx = blockIdx.x * blockDim.x + threadIdx.x;   // E*E threads
  const int n = idx >> 10;          // output row of Wt (= col of W)
  const int k = idx & 1023;         // output col of Wt (= row of W)
  float acc = W[(size_t)k * E_DIM + n];
  if (la != nullptr) {
    #pragma unroll
    for (int r = 0; r < R_RANK; ++r)
      acc += la[k * R_RANK + r] * lb[r * E_DIM + n];
  }
  outT[idx] = (__bf16)acc;
}

__global__ void f32_to_bf16_k(const float* __restrict__ in, __bf16* __restrict__ out, int n4)
{
  const int i = blockIdx.x * blockDim.x + threadIdx.x;
  if (i < n4) {
    f32x4 v = *(const f32x4*)(in + 4 * (size_t)i);
    bf16x4 o = { (__bf16)v.x, (__bf16)v.y, (__bf16)v.z, (__bf16)v.w };
    *(bf16x4*)(out + 4 * (size_t)i) = o;
  }
}

__global__ void zero_f32_k(float* p, int n)
{
  const int i = blockIdx.x * blockDim.x + threadIdx.x;
  if (i < n) p[i] = 0.0f;
}

// ctx f32 [bh][d][e] -> bf16 TRANSPOSED [bh][e][d]  (B-operand for attn GEMM)
__global__ void ctx_to_bf16_t_k(const float* __restrict__ in, __bf16* __restrict__ outT, int n)
{
  const int i = blockIdx.x * blockDim.x + threadIdx.x;
  if (i < n) {
    const int bh = i >> 12;
    const int e  = (i >> 6) & 63;
    const int d  = i & 63;
    outT[i] = (__bf16)in[((size_t)bh << 12) + (d << 6) + e];
  }
}

// q softmax over head_dim (64 contiguous): one (b,s) row per block, 16 lanes per head
__global__ __launch_bounds__(256)
void softmax_head_k(const float* __restrict__ q, __bf16* __restrict__ out)
{
  const int tid = threadIdx.x;
  const size_t base = (size_t)blockIdx.x * E_DIM + (size_t)(tid >> 4) * D_DIM + (size_t)(tid & 15) * 4;
  f32x4 v = *(const f32x4*)(q + base);
  float m = fmaxf(fmaxf(v.x, v.y), fmaxf(v.z, v.w));
  #pragma unroll
  for (int off = 1; off < 16; off <<= 1)
    m = fmaxf(m, __shfl_xor(m, off, 16));
  float e0 = __expf(v.x - m), e1 = __expf(v.y - m), e2 = __expf(v.z - m), e3 = __expf(v.w - m);
  float s = e0 + e1 + e2 + e3;
  #pragma unroll
  for (int off = 1; off < 16; off <<= 1)
    s += __shfl_xor(s, off, 16);
  const float inv = 1.0f / s;
  bf16x4 o = { (__bf16)(e0 * inv), (__bf16)(e1 * inv), (__bf16)(e2 * inv), (__bf16)(e3 * inv) };
  *(bf16x4*)(out + base) = o;
}

// k softmax over sequence axis (stride E) with key-padding mask; coalesced columns
__global__ __launch_bounds__(256)
void softmax_seq_k(const float* __restrict__ k, const unsigned char* __restrict__ mask,
                   __bf16* __restrict__ out)
{
  const int b = blockIdx.x;
  const int c = blockIdx.y * 256 + threadIdx.x;
  const float* kb = k + (size_t)b * S_LEN * E_DIM + c;
  const unsigned char* mb = mask + (size_t)b * S_LEN;

  float m = -3.4e38f;
  for (int s = 0; s < S_LEN; ++s) {
    float v = mb[s] ? -3.4e38f : kb[(size_t)s * E_DIM];
    m = fmaxf(m, v);
  }
  float sum = 0.0f;
  for (int s = 0; s < S_LEN; ++s)
    if (!mb[s]) sum += __expf(kb[(size_t)s * E_DIM] - m);
  const float inv = 1.0f / sum;

  __bf16* ob = out + (size_t)b * S_LEN * E_DIM + c;
  for (int s = 0; s < S_LEN; ++s) {
    float v = mb[s] ? 0.0f : __expf(kb[(size_t)s * E_DIM] - m) * inv;
    ob[(size_t)s * E_DIM] = (__bf16)v;
  }
}

// ---------------------------------------------------------------------------
// launch
// ---------------------------------------------------------------------------
extern "C" void kernel_launch(void* const* d_in, const int* in_sizes, int n_in,
                              void* d_out, int out_size, void* d_ws, size_t ws_size,
                              hipStream_t stream)
{
  const float* query = (const float*)d_in[0];
  const float* key   = (const float*)d_in[1];
  const float* value = (const float*)d_in[2];
  const unsigned char* mask = (const unsigned char*)d_in[3];
  const float* Wq = (const float*)d_in[4];  const float* bq = (const float*)d_in[5];
  const float* Wk = (const float*)d_in[6];  const float* bk = (const float*)d_in[7];
  const float* Wv = (const float*)d_in[8];  const float* bv = (const float*)d_in[9];
  const float* qla = (const float*)d_in[10]; const float* qlb = (const float*)d_in[11];
  const float* kla = (const float*)d_in[12]; const float* klb = (const float*)d_in[13];
  const float* vla = (const float*)d_in[14]; const float* vlb = (const float*)d_in[15];
  const float* Wo = (const float*)d_in[16]; const float* bo = (const float*)d_in[17];
  float* out = (float*)d_out;

  char* ws = (char*)d_ws;
  size_t off = 0;
  auto alloc = [&](size_t bytes) -> void* {
    void* p = ws + off;
    off += (bytes + 255) & ~(size_t)255;
    return p;
  };

  const size_t ME = (size_t)B_NUM * S_LEN * E_DIM;          // 16,777,216 elems
  const int    NCTX = B_NUM * H_NUM * D_DIM * D_DIM;        // 262,144

  __bf16* q16    = (__bf16*)alloc(ME * 2);
  __bf16* k16    = (__bf16*)alloc(ME * 2);
  __bf16* v16    = (__bf16*)alloc(ME * 2);
  __bf16* Wq16t  = (__bf16*)alloc((size_t)E_DIM * E_DIM * 2);
  __bf16* Wk16t  = (__bf16*)alloc((size_t)E_DIM * E_DIM * 2);
  __bf16* Wv16t  = (__bf16*)alloc((size_t)E_DIM * E_DIM * 2);
  __bf16* Wo16t  = (__bf16*)alloc((size_t)E_DIM * E_DIM * 2);
  float*  qf     = (float*)alloc(ME * 4);
  float*  kf     = (float*)alloc(ME * 4);
  __bf16* vp16   = (__bf16*)alloc(ME * 2);
  __bf16* qn16   = (__bf16*)alloc(ME * 2);
  __bf16* kn16   = (__bf16*)alloc(ME * 2);
  float*  ctxf   = (float*)alloc((size_t)NCTX * 4);
  __bf16* ctx16t = (__bf16*)alloc((size_t)NCTX * 2);
  __bf16* attn16 = (__bf16*)alloc(ME * 2);

  // 1) fold LoRA into bf16 weights, transposed (Wo: plain convert+transpose)
  const int wBlocks = (E_DIM * E_DIM) / 256;
  fold_lora_t_k<<<wBlocks, 256, 0, stream>>>(Wq, qla, qlb, Wq16t);
  fold_lora_t_k<<<wBlocks, 256, 0, stream>>>(Wk, kla, klb, Wk16t);
  fold_lora_t_k<<<wBlocks, 256, 0, stream>>>(Wv, vla, vlb, Wv16t);
  fold_lora_t_k<<<wBlocks, 256, 0, stream>>>(Wo, nullptr, nullptr, Wo16t);

  // 2) cast activations to bf16
  const int n4 = (int)(ME / 4);
  f32_to_bf16_k<<<(n4 + 255) / 256, 256, 0, stream>>>(query, q16, n4);
  f32_to_bf16_k<<<(n4 + 255) / 256, 256, 0, stream>>>(key,   k16, n4);
  f32_to_bf16_k<<<(n4 + 255) / 256, 256, 0, stream>>>(value, v16, n4);

  // 3) projections (M=16384, N=K=1024), 128x128 macro tiles
  dim3 gP(E_DIM / 128, (B_NUM * S_LEN) / 128, 1);
  gemm_bf16_wmma<128, float ><<<gP, 256, 0, stream>>>(q16, E_DIM, Wq16t, E_DIM, qf,   E_DIM, bq, E_DIM, 0, 0, 0, 0, 0);
  gemm_bf16_wmma<128, float ><<<gP, 256, 0, stream>>>(k16, E_DIM, Wk16t, E_DIM, kf,   E_DIM, bk, E_DIM, 0, 0, 0, 0, 0);
  gemm_bf16_wmma<128, __bf16><<<gP, 256, 0, stream>>>(v16, E_DIM, Wv16t, E_DIM, vp16, E_DIM, bv, E_DIM, 0, 0, 0, 0, 0);

  // 4) softmaxes
  softmax_head_k<<<B_NUM * S_LEN, 256, 0, stream>>>(qf, qn16);
  softmax_seq_k<<<dim3(B_NUM, E_DIM / 256), 256, 0, stream>>>(kf, mask, kn16);

  // 5) per-head context matrices (K-split + atomic f32 reduce)
  zero_f32_k<<<(NCTX + 255) / 256, 256, 0, stream>>>(ctxf, NCTX);
  ctx_gemm_wmma<<<dim3(8, B_NUM * H_NUM), 256, 0, stream>>>(kn16, vp16, ctxf);
  ctx_to_bf16_t_k<<<(NCTX + 255) / 256, 256, 0, stream>>>(ctxf, ctx16t, NCTX);

  // 6) out_h = q_norm @ ctx per (b,h): M=4096, N=64, K=64  (BN=64 tiles)
  dim3 gA(1, S_LEN / 128, B_NUM * H_NUM);
  gemm_bf16_wmma<64, __bf16><<<gA, 256, 0, stream>>>(
      qn16, E_DIM, ctx16t, D_DIM, attn16, E_DIM, nullptr, D_DIM,
      (long)S_LEN * E_DIM, (long)D_DIM,
      (long)D_DIM * D_DIM,
      (long)S_LEN * E_DIM, (long)D_DIM);

  // 7) output projection -> f32 d_out
  gemm_bf16_wmma<128, float><<<gP, 256, 0, stream>>>(attn16, E_DIM, Wo16t, E_DIM, out, E_DIM, bo, E_DIM, 0, 0, 0, 0, 0);
}
